// DampedLayer_7550552506850
// MI455X (gfx1250) — matrix operand order repaired and checked by reading, not compiled, not run on hardware
//
#include <hip/hip_runtime.h>
#include <math.h>
#include <stdint.h>

typedef __bf16 bf16;
typedef __attribute__((ext_vector_type(16))) __bf16 v16bf;
typedef __attribute__((ext_vector_type(8)))  float  v8f;

#define P_DIM 256
#define H_DIM 1024
#define B_DIM 8
#define L_DIM 4096
#define BL    (B_DIM * L_DIM)       // 32768
#define N1    (2 * P_DIM)           // 512 (p,t flattened)
#define NCHUNK 32
#define LCHUNK (L_DIM / NCHUNK)     // 128
#define AROW   40                   // padded LDS row stride in bf16 (80B: 16B-aligned, conflict-free)

static_assert(N1 == 512 && NCHUNK == 32, "scan decode uses shifts 9/14");

static __device__ __forceinline__ uint32_t bfbits(float f) {
  uint32_t u = __builtin_bit_cast(uint32_t, f);
  return (u + 0x7FFFu + ((u >> 16) & 1u)) >> 16;   // RNE (prep kernels, off critical path)
}
static __device__ __forceinline__ bf16 f2bf(float f) {
  return __builtin_bit_cast(bf16, (uint16_t)bfbits(f));
}
// pack two floats to two bf16 (round-half-up) in 3 VALU: add, add, v_perm_b32
static __device__ __forceinline__ uint32_t pack_bf16(float lo, float hi) {
  uint32_t ul = __builtin_bit_cast(uint32_t, lo) + 0x8000u;
  uint32_t uh = __builtin_bit_cast(uint32_t, hi) + 0x8000u;
  return __builtin_amdgcn_perm(uh, ul, 0x07060302);  // {uh[31:16], ul[31:16]}
}
// one lane's 16 contiguous bf16 (32B) as two b128 loads
static __device__ __forceinline__ v16bf load16(const bf16* p) {
  union { uint4 q[2]; v16bf v; } u;
  u.q[0] = *(const uint4*)(p);
  u.q[1] = *(const uint4*)(p + 8);
  return u.v;
}

// ---------------------------------------------------------------- params
__global__ void param_kernel(const float* __restrict__ steps_raw,
                             const float* __restrict__ G_raw,
                             const float* __restrict__ A_raw,
                             float* __restrict__ prm) {
  int p = threadIdx.x;
  if (p >= P_DIM) return;
  float st = 1.0f / (1.0f + expf(-steps_raw[p]));
  float G  = fmaxf(G_raw[p], 0.0f);
  float Ad = fmaxf(A_raw[p], 0.0f);
  float S  = 1.0f + st * G;
  float alpha = st * st * Ad / S;
  alpha = 1.99f * tanhf(alpha / 1.99f);
  float m11 = 1.0f / S, m12 = -alpha / st, m21 = st / S, m22 = 1.0f - alpha;
  float f1 = st / S, f2 = st * st / S;
  float a = m11, b = m12, c = m21, d = m22;      // M^LCHUNK
  for (int i = 1; i < LCHUNK; ++i) {
    float na = m11 * a + m12 * c, nb = m11 * b + m12 * d;
    float nc = m21 * a + m22 * c, nd = m21 * b + m22 * d;
    a = na; b = nb; c = nc; d = nd;
  }
  prm[p]           = m11; prm[P_DIM   + p] = m12;
  prm[2*P_DIM + p] = m21; prm[3*P_DIM + p] = m22;
  prm[4*P_DIM + p] = f1;  prm[5*P_DIM + p] = f2;
  prm[6*P_DIM + p] = a;   prm[7*P_DIM + p] = b;
  prm[8*P_DIM + p] = c;   prm[9*P_DIM + p] = d;
}

// ------------------------------------------------- weight repack (transposed: [n][k])
__global__ void prep_Bwt(const float* __restrict__ Bm, bf16* __restrict__ Bwt) {
  int i = blockIdx.x * blockDim.x + threadIdx.x;
  if (i >= N1 * H_DIM) return;
  int n = i / H_DIM, h = i - n * H_DIM;
  int p = n >> 1, t = n & 1;
  Bwt[i] = f2bf(Bm[(p * H_DIM + h) * 2 + t]);
}
__global__ void prep_Cwt(const float* __restrict__ Cm, bf16* __restrict__ Cwt) {
  int i = blockIdx.x * blockDim.x + threadIdx.x;
  if (i >= H_DIM * N1) return;
  float v = Cm[i];
  Cwt[i] = f2bf((i & 1) ? -v : v);
}

// ------------------------------------------------------------ WMMA GEMM
// Out[M,N] = A[M,K](f32 -> bf16 in LDS) * Bwt[N,K](bf16, fragment-contiguous)
// 256 threads = 8 wave32 waves; block tile 128x128; wave tile 32x64 (8 WMMA accs).
// K unrolled by 2: LDS buffers and B-fragment register sets swap roles by
// position (no copies, no inter-WMMA v_mov hazards).
template<int K, int N, bool EPI>
__global__ __launch_bounds__(256)
void gemm_bf16_wmma(const float* __restrict__ A, const bf16* __restrict__ Bwt,
                    float* __restrict__ Out,
                    const float* __restrict__ xAdd, const float* __restrict__ Dv) {
  static_assert((K & 63) == 0, "K must be a multiple of 64");
  __shared__ bf16 As[2][128 * AROW];           // 2 x 10KB double buffer
  const int tid  = threadIdx.x;
  const int lane = tid & 31;
  const int wid  = tid >> 5;
  const int mTile = blockIdx.y * 128;
  const int nTile = blockIdx.x * 128;
  const int wm = (wid & 3) * 32;               // 4 waves down M
  const int wn = (wid >> 2) * 64;              // 2 waves across N

  // ---- staging decode: each thread moves 4 float4 (rows sr+32i, cols sk..sk+3)
  const int sr  = tid >> 3;                    // 0..31
  const int sk  = (tid & 7) * 4;               // 0..28, 4-aligned
  const int skh = sk >> 4;
  const int se  = sk & 15;                     // contiguous k (k = 16h + s)
  const float* aSrc = A + (size_t)(mTile + sr) * K + sk;

  auto stage = [&](int buf, int kk) {
    #pragma unroll
    for (int i = 0; i < 4; ++i) {
      const float4 v = *(const float4*)(aSrc + (size_t)i * 32 * K + kk);
      uint32_t p0 = pack_bf16(v.x, v.y);
      uint32_t p1 = pack_bf16(v.z, v.w);
      uint32_t* d = (uint32_t*)(&As[buf][(sr + i * 32) * AROW + skh * 16 + se]);
      d[0] = p0; d[1] = p1;                    // 8B store, 8B-aligned
    }
    if (kk + 64 < K)                           // global_prefetch_b8 of tile after next
      __builtin_prefetch(aSrc + (size_t)32 * K + kk + 64, 0, 1);
  };

  // ---- B fragment base pointer (direct from global: 32B contiguous per lane)
  const int rowL = lane & 15;
  const int kh16 = (lane >> 4) * 16;
  const bf16* bp = Bwt + (size_t)(nTile + wn + rowL) * K + kh16;
  const bf16* aLds0 = &As[0][(wm + rowL) * AROW + kh16];
  const bf16* aLds1 = &As[1][(wm + rowL) * AROW + kh16];

  v8f acc[2][4] = {};
  v16bf b0[4], b1[4];
  #pragma unroll
  for (int ni = 0; ni < 4; ++ni)
    b0[ni] = load16(bp + (size_t)ni * 16 * K);     // prologue B for k=0
  stage(0, 0);

  for (int kk = 0; kk < K; kk += 64) {
    // ================= half 1: compute[b0, As0]; load b1 @kk+32; stage As1 @kk+32
    __syncthreads();
    #pragma unroll
    for (int ni = 0; ni < 4; ++ni)
      b1[ni] = load16(bp + (size_t)ni * 16 * K + kk + 32);
    stage(1, kk + 32);

    {
      v16bf a0 = load16(aLds0);                // 2x ds_load_b128 each
      v16bf a1 = load16(aLds0 + 16 * AROW);
      #pragma unroll
      for (int ni = 0; ni < 4; ++ni) {
        acc[0][ni] = __builtin_amdgcn_wmma_f32_16x16x32_bf16(false, a0, false, b0[ni], (short)0, acc[0][ni], false, false);
        acc[1][ni] = __builtin_amdgcn_wmma_f32_16x16x32_bf16(false, a1, false, b0[ni], (short)0, acc[1][ni], false, false);
      }
    }

    // ================= half 2: compute[b1, As1]; load b0 @kk+64; stage As0 @kk+64
    __syncthreads();
    if (kk + 64 < K) {
      #pragma unroll
      for (int ni = 0; ni < 4; ++ni)
        b0[ni] = load16(bp + (size_t)ni * 16 * K + kk + 64);
      stage(0, kk + 64);
    }

    {
      v16bf a0 = load16(aLds1);
      v16bf a1 = load16(aLds1 + 16 * AROW);
      #pragma unroll
      for (int ni = 0; ni < 4; ++ni) {
        acc[0][ni] = __builtin_amdgcn_wmma_f32_16x16x32_bf16(false, a0, false, b1[ni], (short)0, acc[0][ni], false, false);
        acc[1][ni] = __builtin_amdgcn_wmma_f32_16x16x32_bf16(false, a1, false, b1[ni], (short)0, acc[1][ni], false, false);
      }
    }
  }

  // ---- writeback: f32 C/D layout: VGPR r -> M=r (lanes 0-15) / M=r+8 (16-31)
  const int rhi = (lane >> 4) * 8;
  #pragma unroll
  for (int mi = 0; mi < 2; ++mi) {
    #pragma unroll
    for (int ni = 0; ni < 4; ++ni) {
      int col = nTile + wn + ni * 16 + rowL;
      #pragma unroll
      for (int r = 0; r < 8; ++r) {
        int row = mTile + wm + mi * 16 + r + rhi;
        float v = acc[mi][ni][r];
        if (EPI) v += xAdd[(size_t)row * N + col] * Dv[col];
        Out[(size_t)row * N + col] = v;
      }
    }
  }
}

// --------------------------------------------------------------- scan
__global__ void scan_pass1(const float* __restrict__ Bu, const float* __restrict__ prm,
                           float2* __restrict__ cs) {
  int gid = blockIdx.x * blockDim.x + threadIdx.x;
  if (gid >= B_DIM * NCHUNK * N1) return;
  int n = gid & (N1 - 1);
  int c = (gid >> 9) & (NCHUNK - 1);
  int b = gid >> 14;
  int p = n >> 1;
  float m11 = prm[p],           m12 = prm[P_DIM + p];
  float m21 = prm[2*P_DIM + p], m22 = prm[3*P_DIM + p];
  float f1  = prm[4*P_DIM + p], f2  = prm[5*P_DIM + p];
  size_t base = ((size_t)b * L_DIM + (size_t)c * LCHUNK) * N1 + n;
  float s1 = 0.f, s2 = 0.f;
  for (int l = 0; l < LCHUNK; ++l) {
    float u  = Bu[base + (size_t)l * N1];
    float t1 = m11 * s1 + m12 * s2 + f1 * u;
    float t2 = m21 * s1 + m22 * s2 + f2 * u;
    s1 = t1; s2 = t2;
  }
  cs[gid] = make_float2(s1, s2);
}

__global__ void scan_pass2(const float* __restrict__ prm, const float2* __restrict__ cs,
                           float2* __restrict__ is) {
  int gid = blockIdx.x * blockDim.x + threadIdx.x;
  if (gid >= B_DIM * N1) return;
  int n = gid & (N1 - 1);
  int b = gid >> 9;
  int p = n >> 1;
  float ca = prm[6*P_DIM + p], cb = prm[7*P_DIM + p];
  float cc = prm[8*P_DIM + p], cd = prm[9*P_DIM + p];
  float s1 = 0.f, s2 = 0.f;
  for (int c = 0; c < NCHUNK; ++c) {
    int idx = (b * NCHUNK + c) * N1 + n;
    is[idx] = make_float2(s1, s2);
    float2 d = cs[idx];
    float t1 = ca * s1 + cb * s2 + d.x;
    float t2 = cc * s1 + cd * s2 + d.y;
    s1 = t1; s2 = t2;
  }
}

__global__ void scan_pass3(float* __restrict__ Bu, const float* __restrict__ prm,
                           const float2* __restrict__ is) {
  int gid = blockIdx.x * blockDim.x + threadIdx.x;
  if (gid >= B_DIM * NCHUNK * N1) return;
  int n = gid & (N1 - 1);
  int c = (gid >> 9) & (NCHUNK - 1);
  int b = gid >> 14;
  int p = n >> 1;
  float m11 = prm[p],           m12 = prm[P_DIM + p];
  float m21 = prm[2*P_DIM + p], m22 = prm[3*P_DIM + p];
  float f1  = prm[4*P_DIM + p], f2  = prm[5*P_DIM + p];
  size_t base = ((size_t)b * L_DIM + (size_t)c * LCHUNK) * N1 + n;
  float2 s0 = is[gid];
  float s1 = s0.x, s2 = s0.y;
  for (int l = 0; l < LCHUNK; ++l) {
    size_t a = base + (size_t)l * N1;
    float u  = Bu[a];
    float t1 = m11 * s1 + m12 * s2 + f1 * u;
    float t2 = m21 * s1 + m22 * s2 + f2 * u;
    s1 = t1; s2 = t2;
    Bu[a] = s2;                                // ys in place
  }
}

// --------------------------------------------------------------- launch
extern "C" void kernel_launch(void* const* d_in, const int* in_sizes, int n_in,
                              void* d_out, int out_size, void* d_ws, size_t ws_size,
                              hipStream_t stream) {
  const float* x         = (const float*)d_in[0];
  const float* steps_raw = (const float*)d_in[1];
  const float* G_raw     = (const float*)d_in[2];
  const float* A_raw     = (const float*)d_in[3];
  const float* B_mat     = (const float*)d_in[4];
  const float* C_mat     = (const float*)d_in[5];
  const float* D_vec     = (const float*)d_in[6];
  float* out = (float*)d_out;

  char* ws = (char*)d_ws;
  size_t off = 0;
  auto alloc = [&](size_t bytes) -> void* {
    void* pp = ws + off;
    off = (off + bytes + 255) & ~(size_t)255;
    return pp;
  };
  float*  prm = (float*)alloc(10 * P_DIM * sizeof(float));
  bf16*   Bwt = (bf16*)alloc((size_t)N1 * H_DIM * sizeof(bf16));
  bf16*   Cwt = (bf16*)alloc((size_t)H_DIM * N1 * sizeof(bf16));
  float2* cs  = (float2*)alloc((size_t)B_DIM * NCHUNK * N1 * sizeof(float2));
  float2* is  = (float2*)alloc((size_t)B_DIM * NCHUNK * N1 * sizeof(float2));
  float*  Bu  = (float*)alloc((size_t)BL * N1 * sizeof(float));

  hipLaunchKernelGGL(param_kernel, dim3(1), dim3(256), 0, stream,
                     steps_raw, G_raw, A_raw, prm);
  hipLaunchKernelGGL(prep_Bwt, dim3((N1 * H_DIM) / 256), dim3(256), 0, stream, B_mat, Bwt);
  hipLaunchKernelGGL(prep_Cwt, dim3((H_DIM * N1) / 256), dim3(256), 0, stream, C_mat, Cwt);

  // GEMM 1: Bu[BL,512] = x[BL,1024] * B  (K=1024, N=512)
  hipLaunchKernelGGL((gemm_bf16_wmma<H_DIM, N1, false>),
                     dim3(N1 / 128, BL / 128), dim3(256), 0, stream,
                     x, Bwt, Bu, (const float*)nullptr, (const float*)nullptr);

  int scanT = B_DIM * NCHUNK * N1;  // 131072
  hipLaunchKernelGGL(scan_pass1, dim3(scanT / 256), dim3(256), 0, stream, Bu, prm, cs);
  hipLaunchKernelGGL(scan_pass2, dim3((B_DIM * N1) / 256), dim3(256), 0, stream, prm, cs, is);
  hipLaunchKernelGGL(scan_pass3, dim3(scanT / 256), dim3(256), 0, stream, Bu, prm, is);

  // GEMM 2: out[BL,1024] = ys[BL,512] * C' + x*D  (K=512, N=1024)
  hipLaunchKernelGGL((gemm_bf16_wmma<N1, H_DIM, true>),
                     dim3(H_DIM / 128, BL / 128), dim3(256), 0, stream,
                     Bu, Cwt, out, x, D_vec);
}
